// LaneAllocationGAT_39728447488714
// MI455X (gfx1250) — compile-verified, matrix operand order checked by reading, not run on hardware
//
#include <hip/hip_runtime.h>
#include <hip/hip_bf16.h>

// ---------------------------------------------------------------------------
// GATv2 x3 + linear head for MI455X (gfx1250, wave32, WMMA + TDM)
// N=20000 nodes, E=320000 edges (+N self loops), all widths 256, HEADS=2xHID=128
// ---------------------------------------------------------------------------

#define NN     20000
#define EE     320000
#define TE     (EE + NN)      // edges incl. self loops
#define DW     256            // feature width everywhere (D_IN == HEADS*HID)
#define KDIM   256
#define HID    128
#define HEADS  2
#define OUTD   4
#define TM_TILES (NN / 16)    // 1250 M-tiles

typedef __attribute__((ext_vector_type(16))) __bf16 v16bf;
typedef __attribute__((ext_vector_type(8)))  __bf16 v8bf;
typedef __attribute__((ext_vector_type(8)))  float  v8f;
typedef __attribute__((ext_vector_type(4)))  unsigned int u32x4;
typedef __attribute__((ext_vector_type(8)))  int i32x8;
typedef __attribute__((ext_vector_type(4)))  int i32x4;

union FragBF { v16bf v; v8bf h[2]; };

static __device__ __forceinline__ __bf16 cvt_bf16(float f) {
    unsigned u = __float_as_uint(f);
    u += 0x7FFFu + ((u >> 16) & 1u);          // round-to-nearest-even
    unsigned short hs = (unsigned short)(u >> 16);
    __bf16 r;
    __builtin_memcpy(&r, &hs, 2);
    return r;
}

static __device__ __forceinline__ void atomicMaxF32(float* addr, float v) {
    if (v >= 0.f) atomicMax((int*)addr, __float_as_int(v));
    else          atomicMin((unsigned int*)addr, __float_as_uint(v));
}

static __device__ __forceinline__ float lrelu(float v) {
    return v > 0.f ? v : 0.2f * v;
}

// --------------------------- conversion kernels ----------------------------

__global__ void k_f32_to_bf16(const float* __restrict__ src,
                              __bf16* __restrict__ dst, int n) {
    int t = blockIdx.x * blockDim.x + threadIdx.x;
    if (t < n) dst[t] = cvt_bf16(src[t]);
}

// Bt[n*K + k] = W[k*Nn + n]  (transpose so B fragments load contiguously in k)
__global__ void k_w_to_bf16_t(const float* __restrict__ W,
                              __bf16* __restrict__ Bt, int K, int Nn) {
    int t = blockIdx.x * blockDim.x + threadIdx.x;
    if (t < K * Nn) {
        int n = t / K, k = t % K;
        Bt[t] = cvt_bf16(W[k * Nn + n]);
    }
}

// ------------------------------ WMMA GEMM ----------------------------------
// C[M,256] = A[M,256] x B[256,256]; A row-major bf16, Bt = B transposed bf16.
// Block = 8 waves. Block caches a 64-column B strip (Bt rows [tnBase*16,+64),
// 32 KB) in LDS via the Tensor Data Mover; each wave computes a 16x64 C tile:
// one A fragment feeds 4 v_wmma_f32_16x16x32_bf16 per k-step.

__global__ void __launch_bounds__(256)
k_wmma_gemm(const __bf16* __restrict__ A,
            const __bf16* __restrict__ Bt,
            float* __restrict__ C) {
    __shared__ __bf16 sB[64 * KDIM];          // 32 KB: 64 Bt rows (cols of B)

    const int tnBase  = blockIdx.y * 4;       // first 16-wide n-tile of the 4
    const int rowBase = tnBase * 16;          // first Bt row cached in LDS

    // ---- stage B strip into LDS -------------------------------------------
#if defined(__has_builtin)
#  if __has_builtin(__builtin_amdgcn_tensor_load_to_lds) && \
      __has_builtin(__builtin_amdgcn_s_wait_tensorcnt)
#    define USE_TDM 1
#  endif
#endif
#ifdef USE_TDM
    if (threadIdx.x < 32) {                   // wave 0 issues one TDM op
        unsigned long long ga =
            (unsigned long long)(uintptr_t)(Bt + (size_t)rowBase * KDIM);
        unsigned ldsOff = (unsigned)(uintptr_t)(&sB[0]);   // LDS byte offset

        // D# group 0: count=1 | lds_addr | global_addr[56:0] | type=2
        u32x4 g0;
        g0[0] = 0x1u;
        g0[1] = ldsOff;
        g0[2] = (unsigned)(ga & 0xFFFFFFFFu);
        g0[3] = (unsigned)((ga >> 32) & 0x01FFFFFFu) | (2u << 30);

        // D# group 1: data_size=2B; tensor 256x256; tile 256x64; stride 256
        i32x8 g1;
        g1[0] = (int)(1u << 16);                        // data_size=1 (2 bytes)
        g1[1] = (int)((256u & 0xFFFFu) << 16);          // tensor_dim0 lo16
        g1[2] = (int)((256u >> 16) | ((256u & 0xFFFFu) << 16)); // d0 hi | d1 lo
        g1[3] = (int)((256u >> 16) | (256u << 16));     // d1 hi | tile_dim0=256
        g1[4] = (int)64u;                               // tile_dim1=64
        g1[5] = (int)256u;                              // tensor_dim0_stride lo
        g1[6] = (int)(256u << 16);                      // | tensor_dim1_stride lo
        g1[7] = 0;

        i32x4 gz = {0, 0, 0, 0};
#  if defined(__clang_major__) && __clang_major__ >= 23
        i32x8 gz8 = {0, 0, 0, 0, 0, 0, 0, 0};
        __builtin_amdgcn_tensor_load_to_lds(g0, g1, gz, gz, gz8, 0);
#  else
        __builtin_amdgcn_tensor_load_to_lds(g0, g1, gz, gz, 0);
#  endif
        __builtin_amdgcn_s_wait_tensorcnt(0);
    }
#else
    {   // cooperative fallback: 32 KB strip, 16B per thread per step
        const float4* src = (const float4*)(Bt + (size_t)rowBase * KDIM);
        float4* dst = (float4*)sB;
        for (int i = threadIdx.x; i < (64 * KDIM) / 8; i += 256) dst[i] = src[i];
    }
#endif
    __syncthreads();

    const int wv = threadIdx.x >> 5;
    const int tm = blockIdx.x * 8 + wv;
    if (tm >= TM_TILES) return;

    const int lane = threadIdx.x & 31;
    const int hl   = lane >> 4;               // K sub-block select
    const int idx  = lane & 15;               // row (A) / col (B) within tile

    const __bf16* arow  = A + (size_t)(tm * 16 + idx) * KDIM + hl * 8;
    const __bf16* bbase = &sB[0] + (size_t)idx * KDIM + hl * 8;

    v8f acc[4] = {};
    #pragma unroll
    for (int k0 = 0; k0 < KDIM; k0 += 32) {
        FragBF fa;
        fa.h[0] = *(const v8bf*)(arow + k0);
        fa.h[1] = *(const v8bf*)(arow + k0 + 16);
        #pragma unroll
        for (int t = 0; t < 4; ++t) {
            FragBF fb;
            fb.h[0] = *(const v8bf*)(bbase + t * 16 * KDIM + k0);
            fb.h[1] = *(const v8bf*)(bbase + t * 16 * KDIM + k0 + 16);
            acc[t] = __builtin_amdgcn_wmma_f32_16x16x32_bf16(
                false, fa.v, false, fb.v, (short)0, acc[t], false, false);
        }
    }

    // C/D layout: lanes 0-15 -> M=r, lanes 16-31 -> M=r+8; N = lane&15
    float* cbase = C + (size_t)(tm * 16 + hl * 8) * DW + rowBase + idx;
    #pragma unroll
    for (int t = 0; t < 4; ++t)
        #pragma unroll
        for (int r = 0; r < 8; ++r)
            cbase[(size_t)r * DW + t * 16] = acc[t][r];
}

// --------------------------- per-layer init --------------------------------

__global__ void k_layer_init(float* __restrict__ emax, float* __restrict__ denom,
                             float* __restrict__ agg, int n2, int nagg) {
    int t = blockIdx.x * blockDim.x + threadIdx.x;
    if (t < n2) { emax[t] = -__builtin_inff(); denom[t] = 0.f; }
    if (t < nagg) agg[t] = 0.f;
}

// ------------------------------ edge phase ---------------------------------
// One wave per edge; lane L handles channels [L*8, L*8+8); lanes 0-15 = head0.

__global__ void k_edge_scores(const float* __restrict__ xl,
                              const float* __restrict__ xr,
                              const float* __restrict__ att,
                              const long long* __restrict__ ei,
                              float* __restrict__ e_out,
                              float* __restrict__ emax) {
    int wid = blockIdx.x * (blockDim.x >> 5) + (threadIdx.x >> 5);
    if (wid >= TE) return;
    int lane = threadIdx.x & 31;
    long long s, d;
    if (wid < EE) { s = ei[wid]; d = ei[EE + wid]; } else { s = d = wid - EE; }

    const float* pl = xl + (size_t)s * DW + lane * 8;
    const float* pr = xr + (size_t)d * DW + lane * 8;
    const float* pa = att + lane * 8;   // flat [2*128]; channel index == lane*8

    float4 l0 = *(const float4*)pl, l1 = *(const float4*)(pl + 4);
    float4 r0 = *(const float4*)pr, r1 = *(const float4*)(pr + 4);
    float4 a0 = *(const float4*)pa, a1 = *(const float4*)(pa + 4);

    float acc = lrelu(l0.x + r0.x) * a0.x + lrelu(l0.y + r0.y) * a0.y +
                lrelu(l0.z + r0.z) * a0.z + lrelu(l0.w + r0.w) * a0.w +
                lrelu(l1.x + r1.x) * a1.x + lrelu(l1.y + r1.y) * a1.y +
                lrelu(l1.z + r1.z) * a1.z + lrelu(l1.w + r1.w) * a1.w;

    // reduce within each 16-lane half (per head)
    #pragma unroll
    for (int m = 1; m <= 8; m <<= 1) acc += __shfl_xor(acc, m, 32);

    if ((lane & 15) == 0) {
        int hh = lane >> 4;
        e_out[(size_t)wid * HEADS + hh] = acc;
        atomicMaxF32(&emax[(size_t)d * HEADS + hh], acc);
    }
}

__global__ void k_edge_exp(float* __restrict__ e,
                           const float* __restrict__ emax,
                           float* __restrict__ denom,
                           const long long* __restrict__ ei) {
    int t = blockIdx.x * blockDim.x + threadIdx.x;
    if (t >= TE * HEADS) return;
    int edge = t >> 1, hh = t & 1;
    long long d = (edge < EE) ? ei[EE + edge] : (long long)(edge - EE);
    float ex = __expf(e[t] - emax[(size_t)d * HEADS + hh]);
    e[t] = ex;
    atomicAdd(&denom[(size_t)d * HEADS + hh], ex);
}

__global__ void k_edge_aggregate(const float* __restrict__ e,
                                 const float* __restrict__ denom,
                                 const float* __restrict__ xl,
                                 const long long* __restrict__ ei,
                                 float* __restrict__ agg) {
    int wid = blockIdx.x * (blockDim.x >> 5) + (threadIdx.x >> 5);
    if (wid >= TE) return;
    int lane = threadIdx.x & 31;
    long long s, d;
    if (wid < EE) { s = ei[wid]; d = ei[EE + wid]; } else { s = d = wid - EE; }
    int hh = lane >> 4;
    float alpha = e[(size_t)wid * HEADS + hh] / denom[(size_t)d * HEADS + hh];
    const float* pl = xl + (size_t)s * DW + lane * 8;
    float* po = agg + (size_t)d * DW + lane * 8;
    #pragma unroll
    for (int i = 0; i < 8; ++i) atomicAdd(po + i, alpha * pl[i]);
}

__global__ void k_bias_relu(const float* __restrict__ agg,
                            const float* __restrict__ b,
                            float* __restrict__ h, int n) {
    int t = blockIdx.x * blockDim.x + threadIdx.x;
    if (t < n) {
        float v = agg[t] + b[t & (DW - 1)];
        h[t] = v > 0.f ? v : 0.f;
    }
}

// ------------------------------ final head ---------------------------------

__global__ void k_final_linear(const float* __restrict__ h,
                               const float* __restrict__ Wout,
                               const float* __restrict__ bout,
                               float* __restrict__ y) {
    int t = blockIdx.x * blockDim.x + threadIdx.x;
    if (t >= NN * OUTD) return;
    int n = t >> 2, o = t & 3;
    const float* ph = h + (size_t)n * DW;
    float acc = bout[o];
    #pragma unroll 8
    for (int c = 0; c < DW; ++c) acc += ph[c] * Wout[c * OUTD + o];
    y[t] = acc;
}

// ---------------------------------------------------------------------------

extern "C" void kernel_launch(void* const* d_in, const int* in_sizes, int n_in,
                              void* d_out, int out_size, void* d_ws, size_t ws_size,
                              hipStream_t stream) {
    const float*     x  = (const float*)d_in[0];
    const long long* ei = (const long long*)d_in[1];   // int64 edge_index [2,E]
    const float* Wl[3]  = {(const float*)d_in[2], (const float*)d_in[6],  (const float*)d_in[10]};
    const float* Wr[3]  = {(const float*)d_in[3], (const float*)d_in[7],  (const float*)d_in[11]};
    const float* att[3] = {(const float*)d_in[4], (const float*)d_in[8],  (const float*)d_in[12]};
    const float* bia[3] = {(const float*)d_in[5], (const float*)d_in[9],  (const float*)d_in[13]};
    const float* Wout   = (const float*)d_in[14];
    const float* bout   = (const float*)d_in[15];
    float* y = (float*)d_out;

    // workspace carve-out (256B aligned)
    char* ws = (char*)d_ws;
    auto carve = [&](size_t bytes) -> char* {
        char* p = ws;
        ws += (bytes + 255) & ~(size_t)255;
        return p;
    };
    __bf16* hbf   = (__bf16*)carve((size_t)NN * DW * 2);
    __bf16* wlb   = (__bf16*)carve((size_t)DW * DW * 2);
    __bf16* wrb   = (__bf16*)carve((size_t)DW * DW * 2);
    float*  xl    = (float*)carve((size_t)NN * DW * 4);
    float*  xr    = (float*)carve((size_t)NN * DW * 4);
    float*  agg   = (float*)carve((size_t)NN * DW * 4);
    float*  hbuf  = (float*)carve((size_t)NN * DW * 4);
    float*  ebuf  = (float*)carve((size_t)TE * HEADS * 4);
    float*  emax  = (float*)carve((size_t)NN * HEADS * 4);
    float*  denom = (float*)carve((size_t)NN * HEADS * 4);

    const int B = 256;
    const int nFeat  = NN * DW;                          // 5.12M
    const int gFeat  = (nFeat + B - 1) / B;
    const int gW     = (DW * DW + B - 1) / B;
    const dim3 gGemm((TM_TILES + 7) / 8, 4);             // 8 waves x 16x64 tile
    const int gEdgeW = (TE + 7) / 8;                     // 1 wave per edge
    const int gEdgeT = (TE * HEADS + B - 1) / B;
    const int gFinal = (NN * OUTD + B - 1) / B;

    for (int li = 0; li < 3; ++li) {
        const float* feat = (li == 0) ? x : hbuf;

        k_f32_to_bf16<<<gFeat, B, 0, stream>>>(feat, hbf, nFeat);
        k_w_to_bf16_t<<<gW, B, 0, stream>>>(Wl[li], wlb, DW, DW);
        k_w_to_bf16_t<<<gW, B, 0, stream>>>(Wr[li], wrb, DW, DW);

        k_wmma_gemm<<<gGemm, B, 0, stream>>>(hbf, wlb, xl);
        k_wmma_gemm<<<gGemm, B, 0, stream>>>(hbf, wrb, xr);

        k_layer_init<<<gFeat, B, 0, stream>>>(emax, denom, agg, NN * HEADS, nFeat);

        k_edge_scores<<<gEdgeW, B, 0, stream>>>(xl, xr, att[li], ei, ebuf, emax);
        k_edge_exp<<<gEdgeT, B, 0, stream>>>(ebuf, emax, denom, ei);
        k_edge_aggregate<<<gEdgeW, B, 0, stream>>>(ebuf, denom, xl, ei, agg);

        k_bias_relu<<<gFeat, B, 0, stream>>>(agg, bia[li], hbuf, nFeat);
    }

    k_final_linear<<<gFinal, B, 0, stream>>>(hbuf, Wout, bout, y);
}
